// GeoPrediction_64742337020083
// MI455X (gfx1250) — matrix operand order repaired and checked by reading, not compiled, not run on hardware
//
#include <hip/hip_runtime.h>
#include <hip/hip_bf16.h>

typedef __attribute__((ext_vector_type(16))) __bf16 v16bf;
typedef __attribute__((ext_vector_type(8)))  __bf16 v8bf;
typedef __attribute__((ext_vector_type(8)))  float  v8f;
typedef __attribute__((ext_vector_type(4)))  unsigned int u32x4;
typedef __attribute__((ext_vector_type(8)))  int i32x8;
typedef __attribute__((ext_vector_type(4)))  int i32x4;

#define DIM   256
#define HIST  200
#define HT    16
#define NTILE 13            // ceil(HIST/HT)
#define LDS_W1T   0         // 256*256 bf16 = 131072
#define LDS_HF    131072    // 16*256 f32   = 16384
#define LDS_HB    147456    // 16*256 bf16  = 8192
#define LDS_CT    155648    // 16*256 f32   = 16384
#define LDS_RED   172032    // 256 f32      = 1024
#define LDS_TOTAL 173056

// branchless tanh: exact at +/-inf limits, ~1 ulp core (v_exp_f32 + v_rcp_f32)
__device__ __forceinline__ float tanh_fast(float x) {
  float e = __expf(2.f * x);
  return 1.f - 2.f * __builtin_amdgcn_rcpf(e + 1.f);
}

// ---- prep: W1^T in bf16 (w1t[n][k] = W1[k][n]) ----
__global__ void prep_w1t(const float* __restrict__ W1, __bf16* __restrict__ w1t) {
  int t = blockIdx.x * blockDim.x + threadIdx.x;   // 65536 threads
  int n = t >> 8, k = t & 255;
  w1t[t] = (__bf16)W1[k * DIM + n];
}

// ---- prep: C = P @ W2 + bias  (batch-independent, [200,256]) ----
__global__ void prep_c(const float* __restrict__ P, const float* __restrict__ W2,
                       const float* __restrict__ bias, float* __restrict__ cbuf) {
  int t = blockIdx.x * blockDim.x + threadIdx.x;
  if (t >= HIST * DIM) return;
  int h = t >> 8, e = t & 255;
  float acc = bias[t];
  const float* p = P + h * DIM;
  #pragma unroll 8
  for (int d = 0; d < DIM; ++d) acc = fmaf(p[d], W2[d * DIM + e], acc);
  cbuf[t] = acc;
}

// ---- main: one workgroup per batch element ----
__global__ __launch_bounds__(256, 1)
void geo_main(const int* __restrict__ batch_u, const int* __restrict__ batch_v,
              const int* __restrict__ batch_hist,
              const float* __restrict__ user_emb, const float* __restrict__ venue_emb,
              const __bf16* __restrict__ w1t, const float* __restrict__ cbuf,
              const float* __restrict__ a_scalar, float* __restrict__ out) {
  extern __shared__ char smem[];
  __bf16* s_w1t = (__bf16*)(smem + LDS_W1T);
  float*  s_hf  = (float*)(smem + LDS_HF);
  __bf16* s_hb  = (__bf16*)(smem + LDS_HB);
  float*  s_ct  = (float*)(smem + LDS_CT);
  float*  s_red = (float*)(smem + LDS_RED);

  const int b    = blockIdx.x;
  const int tid  = threadIdx.x;
  const int lane = tid & 31;
  const int wv   = tid >> 5;
  const int half = lane >> 4;
  const int ln   = lane & 15;
  const int r    = tid >> 4;       // gather: row in tile
  const int j    = tid & 15;       // gather: 16 threads per row

  // ---- TDM DMA of W1^T (bf16, 128 KB) into LDS; overlaps the first gather ----
  if (wv == 0) {
    unsigned long long ga = (unsigned long long)(const void*)w1t;
    unsigned int lds_addr = (unsigned int)(unsigned long long)(void*)s_w1t;
    u32x4 g0 = { 0x1u, lds_addr, (unsigned int)ga,
                 (unsigned int)((ga >> 32) & 0x01FFFFFFu) | 0x80000000u };
    i32x8 g1 = { (int)(3u << 16),        // data_size = 8 bytes
                 (int)(16384u << 16),    // tensor_dim0[15:0] << 16
                 (int)(1u << 16),        // tensor_dim0 hi | tensor_dim1 = 1
                 (int)(16384u << 16),    // tile_dim0 = 16384
                 0, 16384, 0, 0 };       // tensor_dim0_stride = 16384
    i32x4 gz4 = { 0, 0, 0, 0 };
#if defined(__clang_major__) && (__clang_major__ >= 23)
    i32x8 gz8 = { 0, 0, 0, 0, 0, 0, 0, 0 };
    __builtin_amdgcn_tensor_load_to_lds(g0, g1, gz4, gz4, gz8, 0);
#else
    __builtin_amdgcn_tensor_load_to_lds(g0, g1, gz4, gz4, 0);
#endif
  }

  // softmax/weighted-sum partial sums, per WMMA lane, per nt column-tile.
  // tanh bounds h to [-1,1] -> plain exp, no running max needed.
  float dacc0 = 0.f, nacc0 = 0.f, dacc1 = 0.f, nacc1 = 0.f;
  const int* hrow = batch_hist + b * HIST;

  for (int t = 0; t < NTILE; ++t) {
    // ---- (a) gather 16 history rows, max-norm clamp, stage f32+bf16; copy C tile ----
    int grow = t * HT + r;
    int idx  = (grow < HIST) ? hrow[grow] : 0;
    const float4* src = (const float4*)(venue_emb + (size_t)idx * DIM + j * 16);
    float4 f0 = src[0], f1 = src[1], f2 = src[2], f3 = src[3];
    int prow = grow + HT;   // prefetch next tile's gather rows
    if (prow < HIST)
      __builtin_prefetch(venue_emb + (size_t)hrow[prow] * DIM + j * 16, 0, 0);
    {   // C tile (row clamped for the padded tail; those rows are masked later)
      int cr = (grow < HIST) ? grow : (HIST - 1);
      const float4* cs = (const float4*)(cbuf + cr * DIM + j * 16);
      float4* cd = (float4*)(s_ct + r * DIM + j * 16);
      cd[0] = cs[0]; cd[1] = cs[1]; cd[2] = cs[2]; cd[3] = cs[3];
    }
    float vals[16] = {f0.x, f0.y, f0.z, f0.w, f1.x, f1.y, f1.z, f1.w,
                      f2.x, f2.y, f2.z, f2.w, f3.x, f3.y, f3.z, f3.w};
    float ss = 0.f;
    #pragma unroll
    for (int q = 0; q < 16; ++q) ss += vals[q] * vals[q];
    // row-norm reduction across the 16 lanes of this row (one lane half)
    #pragma unroll
    for (int m = 8; m >= 1; m >>= 1) ss += __shfl_xor(ss, m, 32);
    float nrm = sqrtf(ss);
    float sc  = fminf(1.f, 1.f / fmaxf(nrm, 1e-12f));
    #pragma unroll
    for (int q = 0; q < 16; ++q) {
      float x = vals[q] * sc;
      s_hf[r * DIM + j * 16 + q] = x;
      s_hb[r * DIM + j * 16 + q] = (__bf16)x;
    }
    if (t == 0) {
      // W1^T must have landed in LDS before the first WMMA reads it.
      if (wv == 0) __builtin_amdgcn_s_wait_tensorcnt(0);
    }
    __syncthreads();

    // ---- (b) WMMA (C operand carries P@W2+bias) + fused tanh/softmax epilogue ----
    #pragma unroll
    for (int nt = 0; nt < 2; ++nt) {
      int ncol = (wv * 2 + nt) * 16;
      int col  = ncol + ln;
      v8f acc;
      #pragma unroll
      for (int jv = 0; jv < 8; ++jv)
        acc[jv] = s_ct[(jv + half * 8) * DIM + col];
      #pragma unroll
      for (int kk = 0; kk < 8; ++kk) {
        int kb = kk * 32;
        // A fragment: row M=ln; lane-half K split {0..7,16..23} / {8..15,24..31}
        union { v16bf v; v8bf h[2]; } A, Bf;
        A.h[0] = *(const v8bf*)(s_hb + ln * DIM + kb + half * 8);
        A.h[1] = *(const v8bf*)(s_hb + ln * DIM + kb + 16 + half * 8);
        // B fragment: lane = column n; lane-half K split {0..7,16..23} / {8..15,24..31}
        const __bf16* pb = s_w1t + (ncol + ln) * DIM + kb + half * 8;
        Bf.h[0] = *(const v8bf*)(pb);
        Bf.h[1] = *(const v8bf*)(pb + 16);
        acc = __builtin_amdgcn_wmma_f32_16x16x32_bf16(
            false, A.v, false, Bf.v, (short)0, acc, false, false);
      }
      float dl = 0.f, nl = 0.f;
      #pragma unroll
      for (int jv = 0; jv < 8; ++jv) {
        int row = jv + half * 8;
        int gr  = t * HT + row;
        float p  = __expf(tanh_fast(acc[jv]));
        p = (gr < HIST) ? p : 0.f;           // mask padded rows
        float xv = s_hf[row * DIM + col];
        dl += p;
        nl += p * xv;
      }
      if (nt == 0) { dacc0 += dl; nacc0 += nl; }
      else         { dacc1 += dl; nacc1 += nl; }
    }
    __syncthreads();
  }

  // ---- combine lane-half partials; write l per column ----
  float* s_l = s_hf;   // s_hf no longer needed
  {
    float d0 = dacc0 + __shfl_xor(dacc0, 16, 32);
    float n0 = nacc0 + __shfl_xor(nacc0, 16, 32);
    float d1 = dacc1 + __shfl_xor(dacc1, 16, 32);
    float n1 = nacc1 + __shfl_xor(nacc1, 16, 32);
    if (half == 0) {
      s_l[(wv * 2 + 0) * 16 + ln] = n0 / d0;
      s_l[(wv * 2 + 1) * 16 + ln] = n1 / d1;
    }
  }
  __syncthreads();

  // ---- final scoring (thread tid owns feature column tid) ----
  float le = s_l[tid] + a_scalar[0];
  float ue = user_emb[(size_t)batch_u[b] * DIM + tid];
  float ve = venue_emb[(size_t)batch_v[b] * DIM + tid];

  auto block_sum = [&](float x) -> float {
    s_red[tid] = x;
    __syncthreads();
    for (int s = 128; s > 0; s >>= 1) {
      if (tid < s) s_red[tid] += s_red[tid + s];
      __syncthreads();
    }
    float res = s_red[0];
    __syncthreads();
    return res;
  };

  const float EPS = 1e-12f;
  float un  = sqrtf(block_sum(ue * ue));
  float vn  = sqrtf(block_sum(ve * ve));
  float lnm = sqrtf(block_sum(le * le));
  float s = ue / fmaxf(un, EPS) + le / fmaxf(lnm, EPS) - ve / fmaxf(vn, EPS);
  float tot = block_sum(s * s);
  if (tid == 0) out[b] = sqrtf(tot);
}

extern "C" void kernel_launch(void* const* d_in, const int* in_sizes, int n_in,
                              void* d_out, int out_size, void* d_ws, size_t ws_size,
                              hipStream_t stream) {
  const int*   batch_u    = (const int*)d_in[0];
  const int*   batch_v    = (const int*)d_in[1];
  const int*   batch_hist = (const int*)d_in[2];
  // d_in[3], d_in[4] (olc fields) are read-but-unused by the reference.
  const float* user_emb   = (const float*)d_in[5];
  const float* venue_emb  = (const float*)d_in[6];
  const float* W1   = (const float*)d_in[7];
  const float* W2   = (const float*)d_in[8];
  const float* P    = (const float*)d_in[9];
  const float* bias = (const float*)d_in[10];
  const float* a    = (const float*)d_in[11];
  float* out = (float*)d_out;

  __bf16* w1t  = (__bf16*)d_ws;
  float*  cbuf = (float*)((char*)d_ws + (size_t)DIM * DIM * 2);

  prep_w1t<<<(DIM * DIM) / 256, 256, 0, stream>>>(W1, w1t);
  prep_c<<<(HIST * DIM + 255) / 256, 256, 0, stream>>>(P, W2, bias, cbuf);

  int B = in_sizes[0];
  geo_main<<<B, 256, LDS_TOTAL, stream>>>(batch_u, batch_v, batch_hist,
                                          user_emb, venue_emb, w1t, cbuf, a, out);
}
